// TempME_2628519985808
// MI455X (gfx1250) — matrix operand order repaired and checked by reading, not compiled, not run on hardware
//
#include <hip/hip_runtime.h>
#include <hip/hip_bf16.h>

// ---------------------------------------------------------------------------
// Types for the gfx1250 WMMA builtin (codegen-confirmed shape):
//   v_wmma_f32_16x16x32_bf16 : D(16x16 f32, 8 VGPR) = A(16x32 bf16) * B(32x16 bf16) + C
// ---------------------------------------------------------------------------
typedef __attribute__((ext_vector_type(16))) __bf16         bf16x16;
typedef __attribute__((ext_vector_type(8)))  float          f32x8;
typedef __attribute__((ext_vector_type(8)))  unsigned short u16x8;

__device__ __forceinline__ unsigned short f2bf(float f) {
    // round-to-nearest-even f32 -> bf16
    unsigned int u = __float_as_uint(f);
    u += 0x7fffu + ((u >> 16) & 1u);
    return (unsigned short)(u >> 16);
}

// ---------------------------------------------------------------------------
// One 16x16 output tile of  D = A @ W^T  using v_wmma_f32_16x16x32_bf16.
//  aT      : LDS tile, 16 rows x aStride bf16 (stored as ushort), K-padded w/ zeros
//  wP      : packed weights in global ws: Npad rows x Kpad bf16, row-major W[n][k]
//  kSteps  : Kpad/32 ; nt : output n-tile index
// A layout (16-bit 16x32): lane m(+16g) holds K = kb*32 + g*8 + i (i<8) and
//                           K = kb*32 + 16 + g*8 + (i-8) (i>=8)   -> two b128 LDS loads
// B layout (16-bit 32x16): lane n(+16g) holds same K as A          -> two b128 loads
// ---------------------------------------------------------------------------
__device__ __forceinline__ f32x8 wmma_tile(const unsigned short* aT, int aStride,
                                           const unsigned short* wP, int Kpad,
                                           int kSteps, int nt) {
    const int lane = threadIdx.x & 31;
    const int g = lane >> 4, m = lane & 15;
    const unsigned short* arow = aT + m * aStride + g * 8;
    const unsigned short* wrow = wP + (size_t)(nt * 16 + m) * Kpad + g * 8;
    f32x8 acc = {0.f, 0.f, 0.f, 0.f, 0.f, 0.f, 0.f, 0.f};
    for (int kb = 0; kb < kSteps; ++kb) {
        union { bf16x16 v; u16x8 h[2]; } a, b;
        a.h[0] = *(const u16x8*)(arow + kb * 32);
        a.h[1] = *(const u16x8*)(arow + kb * 32 + 16);
        b.h[0] = *(const u16x8*)(wrow + kb * 32);
        b.h[1] = *(const u16x8*)(wrow + kb * 32 + 16);
        acc = __builtin_amdgcn_wmma_f32_16x16x32_bf16(false, a.v, false, b.v,
                                                      (short)0, acc, false, false);
    }
    return acc;
}

// ---------------------------------------------------------------------------
// Weight pack: f32 (N,K) row-major -> bf16 (Npad,Kpad) row-major, zero padded.
// ---------------------------------------------------------------------------
__global__ void k_pack(const float* __restrict__ src, unsigned short* __restrict__ dst,
                       int N, int K, int Kpad, int total) {
    int i = blockIdx.x * blockDim.x + threadIdx.x;
    if (i >= total) return;
    int n = i / Kpad, k = i - n * Kpad;
    dst[i] = (n < N && k < K) ? f2bf(src[n * K + k]) : (unsigned short)0;
}

#define BW 64   // W
#define LL 3    // L

// ---------------------------------------------------------------------------
// Kernel 1: event linear + both GINE branches. One wave handles 16 events.
// feat[e][0:64]  = gine(src,tgt),  feat[e][64:128] = gine(tgt,src)
// ---------------------------------------------------------------------------
__global__ __launch_bounds__(32) void k_event_gine(
    const int* __restrict__ node_idx, const int* __restrict__ edge_idx,
    const float* __restrict__ t_records, const float* __restrict__ edge_identify,
    const float* __restrict__ node_embed, const float* __restrict__ edge_embed,
    const float* __restrict__ basis_freq, const float* __restrict__ phase,
    const unsigned short* __restrict__ wEv, const float* __restrict__ bEv,
    const unsigned short* __restrict__ wG1, const float* __restrict__ bG1,
    const unsigned short* __restrict__ wG2, const float* __restrict__ bG2,
    float* __restrict__ feat) {
    __shared__ __attribute__((aligned(16))) unsigned short efT[16 * 352]; // event_f bf16
    __shared__ __attribute__((aligned(16))) float          evT[16 * 176]; // event f32
    __shared__ __attribute__((aligned(16))) float          sT[16 * 172];  // src_f
    __shared__ __attribute__((aligned(16))) float          tT[16 * 172];  // tgt_f
    __shared__ __attribute__((aligned(16))) unsigned short aT[16 * 192];  // GEMM A tile
    __shared__ __attribute__((aligned(16))) unsigned short hT[16 * 64];   // hidden bf16

    const int lane = threadIdx.x;
    const int g = lane >> 4, m = lane & 15;
    const int e0 = blockIdx.x * 16;

    if (lane == 0) {  // gfx1250 prefetch path for L2-resident weights
        __builtin_prefetch(wEv, 0, 0);
        __builtin_prefetch(wG1, 0, 0);
        __builtin_prefetch(wG2, 0, 0);
    }

    // --- build event_f tile (bf16) + gather src/tgt node embeddings (f32) ---
    for (int idx = lane; idx < 16 * 352; idx += 32) {
        int r = idx / 352, c = idx - r * 352;
        int e = e0 + r;
        int b = e / (BW * LL), rem = e - b * (BW * LL);
        int w = rem / LL, l = rem - w * LL;
        int bw = b * BW + w;
        float v;
        if (c < 172) {
            v = edge_embed[(size_t)edge_idx[bw * LL + l] * 172 + c];
        } else if (c < 175) {
            v = edge_identify[(bw * LL + l) * 3 + (c - 172)];
        } else if (c < 347) {
            int d = c - 175;
            float dt = t_records[bw * LL + 2] - t_records[bw * LL + l];
            v = __cosf(dt * basis_freq[d] + phase[d]);
        } else {
            v = 0.f;
        }
        efT[idx] = f2bf(v);
    }
    for (int idx = lane; idx < 16 * 172; idx += 32) {
        int r = idx / 172, c = idx - r * 172;
        int e = e0 + r;
        int b = e / (BW * LL), rem = e - b * (BW * LL);
        int w = rem / LL, l = rem - w * LL;
        int base = (b * BW + w) * (2 * LL);
        sT[idx] = node_embed[(size_t)node_idx[base + 2 * l] * 172 + c];
        tT[idx] = node_embed[(size_t)node_idx[base + 2 * l + 1] * 172 + c];
    }
    __syncthreads();

    // --- event = event_f @ Wev^T + b  (K=352: 11 ksteps, N=176: 11 ntiles) ---
    for (int nt = 0; nt < 11; ++nt) {
        f32x8 acc = wmma_tile(efT, 352, wEv, 352, 11, nt);
#pragma unroll
        for (int r = 0; r < 8; ++r) {
            int row = g * 8 + r, col = nt * 16 + m;
            evT[row * 176 + col] = acc[r] + (col < 172 ? bEv[col] : 0.f);
        }
    }
    __syncthreads();

    // --- two GINE branches ---
    for (int var = 0; var < 2; ++var) {
        const float* S = (var == 0) ? sT : tT;
        const float* T = (var == 0) ? tT : sT;
        for (int idx = lane; idx < 16 * 192; idx += 32) {
            int r = idx / 192, c = idx - r * 192;
            unsigned short h = 0;
            if (c < 172) {
                float msg = T[r * 172 + c] + evT[r * 176 + c];
                msg = msg > 0.f ? msg : 0.f;
                h = f2bf(S[r * 172 + c] + msg);
            }
            aT[idx] = h;
        }
        __syncthreads();
        // h = relu(A @ g1^T + b1): K=192 (6 ksteps), N=64 (4 ntiles)
        for (int nt = 0; nt < 4; ++nt) {
            f32x8 acc = wmma_tile(aT, 192, wG1, 192, 6, nt);
#pragma unroll
            for (int r = 0; r < 8; ++r) {
                int row = g * 8 + r, col = nt * 16 + m;
                float v = acc[r] + bG1[col];
                hT[row * 64 + col] = f2bf(v > 0.f ? v : 0.f);
            }
        }
        __syncthreads();
        // out = h @ g2^T + b2: K=64 (2 ksteps), N=64 -> feat
        for (int nt = 0; nt < 4; ++nt) {
            f32x8 acc = wmma_tile(hT, 64, wG2, 64, 2, nt);
#pragma unroll
            for (int r = 0; r < 8; ++r) {
                int row = g * 8 + r, col = nt * 16 + m;
                feat[(size_t)(e0 + row) * 128 + var * 64 + col] = acc[r] + bG2[col];
            }
        }
        __syncthreads();
    }
}

// ---------------------------------------------------------------------------
// Kernel 2: attention + MLP head. One wave handles 16 windows.
// ---------------------------------------------------------------------------
__global__ __launch_bounds__(32) void k_att_mlp(
    const float* __restrict__ feat, const int* __restrict__ cat_feat,
    const unsigned short* __restrict__ wA1, const float* __restrict__ bA1,
    const unsigned short* __restrict__ wA2, const float* __restrict__ bA2,
    const unsigned short* __restrict__ wM1, const float* __restrict__ bM1,
    const unsigned short* __restrict__ wM2, const float* __restrict__ bM2,
    const unsigned short* __restrict__ wP1, const float* __restrict__ bP1,
    const unsigned short* __restrict__ wP2, const float* __restrict__ bP2,
    const float* __restrict__ w3, const float* __restrict__ b3,
    float* __restrict__ out) {
    __shared__ __attribute__((aligned(16))) float          WpT[16 * 128];
    __shared__ __attribute__((aligned(16))) float          Wq0T[16 * 128];
    __shared__ __attribute__((aligned(16))) float          Wq1T[16 * 128];
    __shared__ __attribute__((aligned(16))) float          srcT[16 * 128];
    __shared__ __attribute__((aligned(16))) unsigned short aT[16 * 128];
    __shared__ __attribute__((aligned(16))) float          hS[16 * 80];
    __shared__ float alphaT[16 * 2];

    const int lane = threadIdx.x;
    const int g = lane >> 4, m = lane & 15;
    const int wd0 = blockIdx.x * 16;

    if (lane == 0) {
        __builtin_prefetch(wA1, 0, 0);
        __builtin_prefetch(wA2, 0, 0);
        __builtin_prefetch(wM1, 0, 0);
    }

    // src rows (feat of event l=2) -> srcT (f32) and A tile (bf16)
    for (int idx = lane; idx < 16 * 128; idx += 32) {
        int r = idx >> 7, c = idx & 127;
        float v = feat[((size_t)(wd0 + r) * 3 + 2) * 128 + c];
        srcT[idx] = v;
        aT[idx] = f2bf(v);
    }
    __syncthreads();
    // Wp = src @ att_w1^T + b  (K=128: 4 ksteps, N=128: 8 ntiles)
    for (int nt = 0; nt < 8; ++nt) {
        f32x8 acc = wmma_tile(aT, 128, wA1, 128, 4, nt);
#pragma unroll
        for (int r = 0; r < 8; ++r) {
            int row = g * 8 + r, col = nt * 16 + m;
            WpT[row * 128 + col] = acc[r] + bA1[col];
        }
    }
    __syncthreads();
    // Wq_k = tgt_k @ att_w2^T + b (k = 0, 1)
    for (int kq = 0; kq < 2; ++kq) {
        for (int idx = lane; idx < 16 * 128; idx += 32) {
            int r = idx >> 7, c = idx & 127;
            aT[idx] = f2bf(feat[((size_t)(wd0 + r) * 3 + kq) * 128 + c]);
        }
        __syncthreads();
        float* dst = kq ? Wq1T : Wq0T;
        for (int nt = 0; nt < 8; ++nt) {
            f32x8 acc = wmma_tile(aT, 128, wA2, 128, 4, nt);
#pragma unroll
            for (int r = 0; r < 8; ++r) {
                int row = g * 8 + r, col = nt * 16 + m;
                dst[row * 128 + col] = acc[r] + bA2[col];
            }
        }
        __syncthreads();
    }
    // scores + 2-way softmax (one window per lane)
    if (lane < 16) {
        float s0 = 0.f, s1 = 0.f;
        for (int c = 0; c < 128; ++c) {
            float p = WpT[lane * 128 + c];
            s0 += p * Wq0T[lane * 128 + c];
            s1 += p * Wq1T[lane * 128 + c];
        }
        float mx = fmaxf(s0, s1);
        float e0 = __expf(s0 - mx), e1 = __expf(s1 - mx);
        float inv = 1.f / (e0 + e1);
        alphaT[lane * 2 + 0] = e0 * inv;
        alphaT[lane * 2 + 1] = e1 * inv;
    }
    __syncthreads();
    // out = src + a0*Wq0 + a1*Wq1 -> A tile
    for (int idx = lane; idx < 16 * 128; idx += 32) {
        int r = idx >> 7;
        float v = srcT[idx] + alphaT[r * 2] * Wq0T[idx] + alphaT[r * 2 + 1] * Wq1T[idx];
        aT[idx] = f2bf(v);
    }
    __syncthreads();
    // h1 = relu(out @ m1^T + b): K=128, N=64
    for (int nt = 0; nt < 4; ++nt) {
        f32x8 acc = wmma_tile(aT, 128, wM1, 128, 4, nt);
#pragma unroll
        for (int r = 0; r < 8; ++r) {
            int row = g * 8 + r, col = nt * 16 + m;
            float v = acc[r] + bM1[col];
            hS[row * 64 + col] = v > 0.f ? v : 0.f;
        }
    }
    __syncthreads();
    for (int idx = lane; idx < 16 * 64; idx += 32) aT[idx] = f2bf(hS[idx]);
    __syncthreads();
    // h = h1 @ m2^T + b: K=64, N=64 (no relu)
    for (int nt = 0; nt < 4; ++nt) {
        f32x8 acc = wmma_tile(aT, 64, wM2, 64, 2, nt);
#pragma unroll
        for (int r = 0; r < 8; ++r) {
            int row = g * 8 + r, col = nt * 16 + m;
            hS[row * 64 + col] = acc[r] + bM2[col];
        }
    }
    __syncthreads();
    // x = [h (64) | one_hot(cat,12) | pad] -> 16x96 A tile
    for (int idx = lane; idx < 16 * 96; idx += 32) {
        int r = idx / 96, c = idx - r * 96;
        float v = 0.f;
        if (c < 64) v = hS[r * 64 + c];
        else if (c < 76) v = (cat_feat[wd0 + r] == (c - 64)) ? 1.f : 0.f;
        aT[idx] = f2bf(v);
    }
    __syncthreads();
    // h1 = relu(x @ mlp1^T + b): K=96 (3), N=80 (76 valid)
    for (int nt = 0; nt < 5; ++nt) {
        f32x8 acc = wmma_tile(aT, 96, wP1, 96, 3, nt);
#pragma unroll
        for (int r = 0; r < 8; ++r) {
            int row = g * 8 + r, col = nt * 16 + m;
            if (col < 76) {
                float v = acc[r] + bP1[col];
                hS[row * 80 + col] = v > 0.f ? v : 0.f;
            }
        }
    }
    __syncthreads();
    for (int idx = lane; idx < 16 * 96; idx += 32) {
        int r = idx / 96, c = idx - r * 96;
        aT[idx] = (c < 76) ? f2bf(hS[r * 80 + c]) : (unsigned short)0;
    }
    __syncthreads();
    // h2 = relu(h1 @ mlp2^T + b): K=96 (3), N=64
    for (int nt = 0; nt < 4; ++nt) {
        f32x8 acc = wmma_tile(aT, 96, wP2, 96, 3, nt);
#pragma unroll
        for (int r = 0; r < 8; ++r) {
            int row = g * 8 + r, col = nt * 16 + m;
            float v = acc[r] + bP2[col];
            hS[row * 64 + col] = v > 0.f ? v : 0.f;
        }
    }
    __syncthreads();
    // z = h2 . w3 + b3 -> sigmoid
    if (lane < 16) {
        float z = 0.f;
        for (int c = 0; c < 64; ++c) z += hS[lane * 64 + c] * w3[c];
        z += b3[0];
        out[wd0 + lane] = 1.f / (1.f + __expf(-z));
    }
}

// ---------------------------------------------------------------------------
extern "C" void kernel_launch(void* const* d_in, const int* in_sizes, int n_in,
                              void* d_out, int out_size, void* d_ws, size_t ws_size,
                              hipStream_t stream) {
    (void)in_sizes; (void)n_in; (void)out_size; (void)ws_size;
    const int*   node_idx      = (const int*)d_in[0];
    const int*   edge_idx      = (const int*)d_in[1];
    const int*   cat_feat      = (const int*)d_in[2];
    const float* t_records     = (const float*)d_in[3];
    const float* edge_identify = (const float*)d_in[4];
    /* d_in[5] cut_time_l: unused by the reference */
    const float* node_embed    = (const float*)d_in[6];
    const float* edge_embed    = (const float*)d_in[7];
    const float* basis_freq    = (const float*)d_in[8];
    const float* phase         = (const float*)d_in[9];
    const float* lin_event_w   = (const float*)d_in[10];
    const float* lin_event_b   = (const float*)d_in[11];
    const float* gcn_w1        = (const float*)d_in[12];
    const float* gcn_b1        = (const float*)d_in[13];
    const float* gcn_w2        = (const float*)d_in[14];
    const float* gcn_b2        = (const float*)d_in[15];
    const float* att_w1_w      = (const float*)d_in[16];
    const float* att_w1_b      = (const float*)d_in[17];
    const float* att_w2_w      = (const float*)d_in[18];
    const float* att_w2_b      = (const float*)d_in[19];
    const float* att_m1_w      = (const float*)d_in[20];
    const float* att_m1_b      = (const float*)d_in[21];
    const float* att_m2_w      = (const float*)d_in[22];
    const float* att_m2_b      = (const float*)d_in[23];
    const float* mlp_w1        = (const float*)d_in[24];
    const float* mlp_b1        = (const float*)d_in[25];
    const float* mlp_w2        = (const float*)d_in[26];
    const float* mlp_b2        = (const float*)d_in[27];
    const float* mlp_w3        = (const float*)d_in[28];
    const float* mlp_b3        = (const float*)d_in[29];
    float* out = (float*)d_out;

    char* ws = (char*)d_ws;
    size_t off = 0;
    auto carve = [&](size_t bytes) { size_t o = off; off += (bytes + 127) & ~(size_t)127; return o; };

    const size_t oEv = carve((size_t)176 * 352 * 2);
    const size_t oG1 = carve((size_t)64 * 192 * 2);
    const size_t oG2 = carve((size_t)64 * 64 * 2);
    const size_t oA1 = carve((size_t)128 * 128 * 2);
    const size_t oA2 = carve((size_t)128 * 128 * 2);
    const size_t oM1 = carve((size_t)64 * 128 * 2);
    const size_t oM2 = carve((size_t)64 * 64 * 2);
    const size_t oP1 = carve((size_t)80 * 96 * 2);
    const size_t oP2 = carve((size_t)64 * 96 * 2);
    const size_t oFeat = carve((size_t)98304 * 128 * 4);

    auto pack = [&](const float* src, size_t o, int N, int K, int Kpad, int Npad) {
        int total = Npad * Kpad;
        k_pack<<<(total + 255) / 256, 256, 0, stream>>>(src, (unsigned short*)(ws + o),
                                                        N, K, Kpad, total);
    };
    pack(lin_event_w, oEv, 172, 347, 352, 176);
    pack(gcn_w1,      oG1,  64, 172, 192,  64);
    pack(gcn_w2,      oG2,  64,  64,  64,  64);
    pack(att_w1_w,    oA1, 128, 128, 128, 128);
    pack(att_w2_w,    oA2, 128, 128, 128, 128);
    pack(att_m1_w,    oM1,  64, 128, 128,  64);
    pack(att_m2_w,    oM2,  64,  64,  64,  64);
    pack(mlp_w1,      oP1,  76,  76,  96,  80);
    pack(mlp_w2,      oP2,  64,  76,  96,  64);

    float* feat = (float*)(ws + oFeat);

    // 98304 events / 16 per wave
    k_event_gine<<<6144, 32, 0, stream>>>(
        node_idx, edge_idx, t_records, edge_identify, node_embed, edge_embed,
        basis_freq, phase,
        (const unsigned short*)(ws + oEv), lin_event_b,
        (const unsigned short*)(ws + oG1), gcn_b1,
        (const unsigned short*)(ws + oG2), gcn_b2,
        feat);

    // 32768 windows / 16 per wave
    k_att_mlp<<<2048, 32, 0, stream>>>(
        feat, cat_feat,
        (const unsigned short*)(ws + oA1), att_w1_b,
        (const unsigned short*)(ws + oA2), att_w2_b,
        (const unsigned short*)(ws + oM1), att_m1_b,
        (const unsigned short*)(ws + oM2), att_m2_b,
        (const unsigned short*)(ws + oP1), mlp_b1,
        (const unsigned short*)(ws + oP2), mlp_b2,
        mlp_w3, mlp_b3,
        out);
}
